// CascadedVMambaBlock_38276748541968
// MI455X (gfx1250) — compile-verified, matrix operand order checked by reading, not run on hardware
//
#include <hip/hip_runtime.h>
#include <hip/hip_bf16.h>
#include <cmath>

typedef __attribute__((ext_vector_type(16))) _Float16 v16h;
typedef __attribute__((ext_vector_type(8)))  float    v8f;

constexpr int BATCH = 2, HIMG = 48, WIMG = 48;
constexpr int LSEQ  = HIMG * WIMG;        // 2304
constexpr int BL    = BATCH * LSEQ;       // 4608
constexpr int CIN   = 128;
constexpr int CH    = 32;                 // per-head channels
constexpr int DI    = 64;                 // d_inner
constexpr int DSTATE = 16, DTRANK = 2, KDIR = 4;
constexpr int CDBL  = DTRANK + 2 * DSTATE; // 34
constexpr int CPAD  = 48;                  // x_proj N padded to 3 WMMA tiles
constexpr int NCH   = 48;                  // scan chunks
constexpr int LC    = LSEQ / NCH;          // 48 steps per chunk
constexpr int STATES = DI * DSTATE;        // 1024 states per (b,k)

// -------- WMMA 16-bit operand striping (CDNA5 ISA 7.12.2) ----------------
__device__ __forceinline__ int wmma_k16(int e, int laneHi) {
  int g = e >> 1, p = e & 1;
  int k = (g < 4) ? (2 * g + p) : (16 + 2 * (g - 4) + p);
  return k + laneHi * 8;
}

__device__ __forceinline__ int srcpos(int k, int t) {
  int tt = (k >= 2) ? (LSEQ - 1 - t) : t;
  if (k & 1) { int h = tt % HIMG; int w = tt / HIMG; return h * WIMG + w; }
  return tt;
}

__device__ __forceinline__ float sigmoidf_(float x) { return 1.f / (1.f + __expf(-x)); }

// ========================================================================
// 0) channel shuffle: c' = cg*4 + g  <-  c = g*32 + cg
__global__ void k_shuffle(const float* __restrict__ x, float* __restrict__ x_sh) {
  int idx = blockIdx.x * blockDim.x + threadIdx.x;
  if (idx >= BL * CIN) return;
  int c = idx % CIN, p = idx / CIN;
  int g = c % 4, cg = c / 4;
  x_sh[idx] = x[p * CIN + g * CH + cg];
}

// 0b) one-shot f16 weight pack (x_proj N padded 34 -> 48 with zeros)
__global__ void k_prep_weights(const float* __restrict__ Wip, const float* __restrict__ Wxp,
                               const float* __restrict__ Wop, _Float16* __restrict__ Wip_h,
                               _Float16* __restrict__ Wxp_h, _Float16* __restrict__ Wop_h) {
  int idx = blockIdx.x * blockDim.x + threadIdx.x;
  if (idx < CIN * CH) Wip_h[idx] = (_Float16)Wip[idx];
  if (idx < CH * DI)  Wop_h[idx] = (_Float16)Wop[idx];
  if (idx < KDIR * CPAD * DI) {
    int kk = idx % DI;
    int c  = (idx / DI) % CPAD;
    int k  = idx / (DI * CPAD);
    Wxp_h[idx] = (c < CDBL) ? (_Float16)Wxp[((size_t)k * CDBL + c) * DI + kk] : (_Float16)0.f;
  }
}

// 1) s = chunk + (first ? 0 : prev)
__global__ void k_chunk_input(const float* __restrict__ x_sh, const float* __restrict__ prev,
                              float* __restrict__ s, int chunk, int first) {
  int idx = blockIdx.x * blockDim.x + threadIdx.x;
  if (idx >= BL * CH) return;
  int c = idx % CH, p = idx / CH;
  float v = x_sh[p * CIN + chunk * CH + c];
  if (!first) v += prev[idx];
  s[idx] = v;
}

// 2) LN(32) + in_proj GEMM: (16 pixels x 32) x (32 x 128) via 8 waves/block
__global__ void __launch_bounds__(256) k_ln_inproj(
    const float* __restrict__ s, const float* __restrict__ ln_w,
    const float* __restrict__ ln_b, const _Float16* __restrict__ Wip_h,
    float* __restrict__ xz) {
  __shared__ _Float16 lnA[16][40];
  int row0 = blockIdx.x * 16;
  int tid = threadIdx.x;
  if (tid < 16) {
    const float* r = s + (size_t)(row0 + tid) * CH;
    float m = 0.f;
    #pragma unroll
    for (int c = 0; c < CH; ++c) m += r[c];
    m *= (1.f / CH);
    float v = 0.f;
    #pragma unroll
    for (int c = 0; c < CH; ++c) { float d = r[c] - m; v += d * d; }
    float rs = rsqrtf(v * (1.f / CH) + 1e-5f);
    #pragma unroll
    for (int c = 0; c < CH; ++c)
      lnA[tid][c] = (_Float16)((r[c] - m) * rs * ln_w[c] + ln_b[c]);
  }
  __syncthreads();
  int lane = tid & 31, wave = tid >> 5;
  int laneHi = lane >> 4, ll = lane & 15;
  int n0 = wave * 16;
  v16h a{}, bf{};
  #pragma unroll
  for (int e = 0; e < 16; ++e) {
    int kk = wmma_k16(e, laneHi);
    a[e]  = lnA[ll][kk];
    bf[e] = Wip_h[(size_t)(n0 + ll) * CH + kk];
  }
  v8f acc = {};
  acc = __builtin_amdgcn_wmma_f32_16x16x32_f16(false, a, false, bf, (short)0, acc, false, false);
  #pragma unroll
  for (int r = 0; r < 8; ++r) {
    int row = row0 + r + laneHi * 8;
    xz[(size_t)row * CIN + n0 + ll] = acc[r];
  }
}

// 3) depthwise 3x3 SAME conv + bias + SiLU on xx = xz[:, :64]
__global__ void k_conv_silu(const float* __restrict__ xz, const float* __restrict__ cw,
                            const float* __restrict__ cb, float* __restrict__ xxc) {
  int idx = blockIdx.x * blockDim.x + threadIdx.x;
  if (idx >= BL * DI) return;
  int c = idx % DI, p = idx / DI;
  int b = p / LSEQ, pl = p % LSEQ;
  int h = pl / WIMG, w = pl % WIMG;
  float acc = cb[c];
  #pragma unroll
  for (int ky = 0; ky < 3; ++ky) {
    int ih = h + ky - 1;
    if (ih < 0 || ih >= HIMG) continue;
    #pragma unroll
    for (int kx = 0; kx < 3; ++kx) {
      int iw = w + kx - 1;
      if (iw < 0 || iw >= WIMG) continue;
      acc += xz[((size_t)(b * LSEQ + ih * WIMG + iw)) * CIN + c] * cw[(ky * 3 + kx) * DI + c];
    }
  }
  xxc[idx] = acc * sigmoidf_(acc);
}

// 4) x_proj: per (b,k) 16-step tile, K=64 chained WMMA, 3 N-tiles (padded)
__global__ void __launch_bounds__(32) k_xproj(
    const float* __restrict__ xxc, const _Float16* __restrict__ Wxp_h,
    float* __restrict__ dtr, float* __restrict__ Bs, float* __restrict__ Cs) {
  __shared__ _Float16 As[16][72];
  int tile = blockIdx.x;
  int mt = tile % (LSEQ / 16);
  int bk = tile / (LSEQ / 16);
  int k = bk % KDIR, b = bk / KDIR;
  int t0 = mt * 16;
  int lane = threadIdx.x;
  for (int r = 0; r < 16; ++r) {
    int src = srcpos(k, t0 + r);
    const float* row = xxc + (size_t)(b * LSEQ + src) * DI;
    for (int c = lane; c < DI; c += 32) As[r][c] = (_Float16)row[c];
  }
  __syncthreads();
  int laneHi = lane >> 4, ll = lane & 15;
  v16h a0{}, a1{};
  #pragma unroll
  for (int e = 0; e < 16; ++e) {
    int kk = wmma_k16(e, laneHi);
    a0[e] = As[ll][kk];
    a1[e] = As[ll][kk + 32];
  }
  for (int nt = 0; nt < 3; ++nt) {
    int n0 = nt * 16;
    v16h b0{}, b1{};
    const _Float16* wr = Wxp_h + ((size_t)k * CPAD + n0 + ll) * DI;
    #pragma unroll
    for (int e = 0; e < 16; ++e) {
      int kk = wmma_k16(e, laneHi);
      b0[e] = wr[kk];
      b1[e] = wr[kk + 32];
    }
    v8f acc = {};
    acc = __builtin_amdgcn_wmma_f32_16x16x32_f16(false, a0, false, b0, (short)0, acc, false, false);
    acc = __builtin_amdgcn_wmma_f32_16x16x32_f16(false, a1, false, b1, (short)0, acc, false, false);
    #pragma unroll
    for (int r = 0; r < 8; ++r) {
      int t = t0 + r + laneHi * 8;
      int cc = n0 + ll;
      size_t base = (size_t)bk * LSEQ + t;
      if (cc < DTRANK)               dtr[base * DTRANK + cc] = acc[r];
      else if (cc < DTRANK + DSTATE) Bs[base * DSTATE + (cc - DTRANK)] = acc[r];
      else if (cc < CDBL)            Cs[base * DSTATE + (cc - DTRANK - DSTATE)] = acc[r];
    }
  }
}

// 5) dt_proj (rank 2) + bias + softplus
__global__ void k_dtproj(const float* __restrict__ dtr, const float* __restrict__ Wdt,
                         const float* __restrict__ bdt, float* __restrict__ dts) {
  int idx = blockIdx.x * blockDim.x + threadIdx.x;
  if (idx >= BATCH * KDIR * LSEQ * DI) return;
  int d = idx % DI;
  int bkl = idx / DI;
  int k = (bkl / LSEQ) % KDIR;
  const float* wr = Wdt + ((size_t)k * DI + d) * DTRANK;
  float v = dtr[(size_t)bkl * DTRANK] * wr[0] + dtr[(size_t)bkl * DTRANK + 1] * wr[1]
          + bdt[k * DI + d];
  dts[idx] = (v > 20.f) ? v : log1pf(__expf(v));
}

// 6a) chunked scan phase 1: per-chunk (P = prod a, S = local scan from 0)
__global__ void __launch_bounds__(1024) k_scan_chunk(
    const float* __restrict__ xxc, const float* __restrict__ dts,
    const float* __restrict__ Bs, const float* __restrict__ A_logs,
    float* __restrict__ scanP, float* __restrict__ scanS) {
  int blk = blockIdx.x;          // bk*NCH + c
  int c = blk % NCH, bk = blk / NCH;
  int k = bk % KDIR, b = bk / KDIR;
  int t = threadIdx.x;
  int d = t >> 4, n = t & 15;
  float A = -__expf(A_logs[((size_t)k * DI + d) * DSTATE + n]);
  const float* dts_bk = dts + (size_t)bk * LSEQ * DI;
  const float* Bs_bk  = Bs  + (size_t)bk * LSEQ * DSTATE;
  const float* xx_b   = xxc + (size_t)b  * LSEQ * DI;
  int l0 = c * LC;
  float P = 1.f, S = 0.f;
  float xv = xx_b[srcpos(k, l0) * DI + d];
  float dt = dts_bk[(size_t)l0 * DI + d];
  float Bt = Bs_bk[(size_t)l0 * DSTATE + n];
  for (int i = 0; i < LC; ++i) {
    float xv2 = 0.f, dt2 = 0.f, Bt2 = 0.f;
    if (i + 1 < LC) {
      int ln = l0 + i + 1;
      xv2 = xx_b[srcpos(k, ln) * DI + d];
      dt2 = dts_bk[(size_t)ln * DI + d];
      Bt2 = Bs_bk[(size_t)ln * DSTATE + n];
      __builtin_prefetch(&dts_bk[(size_t)(ln + 1 < LSEQ ? ln + 1 : ln) * DI + d], 0, 1);
    }
    float a = __expf(dt * A);
    S = a * S + (dt * xv) * Bt;
    P *= a;
    xv = xv2; dt = dt2; Bt = Bt2;
  }
  scanP[(size_t)blk * STATES + t] = P;
  scanS[(size_t)blk * STATES + t] = S;
}

// 6b) phase 2: tiny serial prefix over chunk summaries -> incoming state I_c
__global__ void __launch_bounds__(1024) k_scan_prefix(
    const float* __restrict__ scanP, const float* __restrict__ scanS,
    float* __restrict__ scanI) {
  int bk = blockIdx.x;
  int t = threadIdx.x;
  float I = 0.f;
  for (int c = 0; c < NCH; ++c) {
    size_t idx = ((size_t)bk * NCH + c) * STATES + t;
    scanI[idx] = I;
    I = scanP[idx] * I + scanS[idx];
  }
}

// 6c) phase 3: replay each chunk from I_c, emit y (16-lane shuffle contraction)
__global__ void __launch_bounds__(1024) k_scan_apply(
    const float* __restrict__ xxc, const float* __restrict__ dts,
    const float* __restrict__ Bs, const float* __restrict__ Cs,
    const float* __restrict__ A_logs, const float* __restrict__ Dsv,
    const float* __restrict__ scanI, float* __restrict__ ys) {
  int blk = blockIdx.x;
  int c = blk % NCH, bk = blk / NCH;
  int k = bk % KDIR, b = bk / KDIR;
  int t = threadIdx.x;
  int d = t >> 4, n = t & 15;
  float A  = -__expf(A_logs[((size_t)k * DI + d) * DSTATE + n]);
  float Dv = Dsv[k * DI + d];
  const float* dts_bk = dts + (size_t)bk * LSEQ * DI;
  const float* Bs_bk  = Bs  + (size_t)bk * LSEQ * DSTATE;
  const float* Cs_bk  = Cs  + (size_t)bk * LSEQ * DSTATE;
  float*       ys_bk  = ys  + (size_t)bk * LSEQ * DI;
  const float* xx_b   = xxc + (size_t)b  * LSEQ * DI;
  int l0 = c * LC;
  float h = scanI[(size_t)blk * STATES + t];
  float xv = xx_b[srcpos(k, l0) * DI + d];
  float dt = dts_bk[(size_t)l0 * DI + d];
  float Bt = Bs_bk[(size_t)l0 * DSTATE + n];
  float Ct = Cs_bk[(size_t)l0 * DSTATE + n];
  for (int i = 0; i < LC; ++i) {
    int l = l0 + i;
    float xv2 = 0.f, dt2 = 0.f, Bt2 = 0.f, Ct2 = 0.f;
    if (i + 1 < LC) {
      int ln = l + 1;
      xv2 = xx_b[srcpos(k, ln) * DI + d];
      dt2 = dts_bk[(size_t)ln * DI + d];
      Bt2 = Bs_bk[(size_t)ln * DSTATE + n];
      Ct2 = Cs_bk[(size_t)ln * DSTATE + n];
    }
    h = __expf(dt * A) * h + (dt * xv) * Bt;
    float p = h * Ct;
    p += __shfl_xor(p, 8, 16);
    p += __shfl_xor(p, 4, 16);
    p += __shfl_xor(p, 2, 16);
    p += __shfl_xor(p, 1, 16);
    if (n == 0) ys_bk[(size_t)l * DI + d] = p + Dv * xv;
    xv = xv2; dt = dt2; Bt = Bt2; Ct = Ct2;
  }
}

// 7) merge 4 directions + out_norm LN(64) * SiLU(z)
__global__ void __launch_bounds__(64) k_combine(
    const float* __restrict__ ys, const float* __restrict__ xz,
    const float* __restrict__ onw, const float* __restrict__ onb,
    float* __restrict__ yact) {
  __shared__ float red[64];
  int p = blockIdx.x;
  int b = p / LSEQ, pl = p % LSEQ;
  int h = pl / WIMG, w = pl % WIMG;
  int d = threadIdx.x;
  size_t base = (size_t)b * KDIR * LSEQ * DI;
  int l1 = w * HIMG + h;
  float y = ys[base + ((size_t)0 * LSEQ + pl) * DI + d]
          + ys[base + ((size_t)2 * LSEQ + (LSEQ - 1 - pl)) * DI + d]
          + ys[base + ((size_t)1 * LSEQ + l1) * DI + d]
          + ys[base + ((size_t)3 * LSEQ + (LSEQ - 1 - l1)) * DI + d];
  red[d] = y; __syncthreads();
  for (int off = 32; off > 0; off >>= 1) { if (d < off) red[d] += red[d + off]; __syncthreads(); }
  float m = red[0] * (1.f / DI); __syncthreads();
  float df = y - m;
  red[d] = df * df; __syncthreads();
  for (int off = 32; off > 0; off >>= 1) { if (d < off) red[d] += red[d + off]; __syncthreads(); }
  float rs = rsqrtf(red[0] * (1.f / DI) + 1e-5f);
  float z = xz[(size_t)p * CIN + DI + d];
  yact[(size_t)p * DI + d] = (df * rs * onw[d] + onb[d]) * (z * sigmoidf_(z));
}

// 8) out_proj GEMM (16x64 x 64x32) + VSS residual: out = gemm + (1+vss_skip)*s
__global__ void __launch_bounds__(32) k_outproj(
    const float* __restrict__ yact, const _Float16* __restrict__ Wop_h,
    const float* __restrict__ s, const float* __restrict__ vss_skip,
    float* __restrict__ prev, float* __restrict__ x_cat, int chunk) {
  __shared__ _Float16 As[16][72];
  int row0 = blockIdx.x * 16;
  int lane = threadIdx.x;
  for (int r = 0; r < 16; ++r) {
    const float* row = yact + (size_t)(row0 + r) * DI;
    for (int c = lane; c < DI; c += 32) As[r][c] = (_Float16)row[c];
  }
  __syncthreads();
  int laneHi = lane >> 4, ll = lane & 15;
  v16h a0{}, a1{};
  #pragma unroll
  for (int e = 0; e < 16; ++e) {
    int kk = wmma_k16(e, laneHi);
    a0[e] = As[ll][kk];
    a1[e] = As[ll][kk + 32];
  }
  float vskip = vss_skip[0];
  for (int nt = 0; nt < 2; ++nt) {
    int n0 = nt * 16;
    v16h b0{}, b1{};
    const _Float16* wr = Wop_h + (size_t)(n0 + ll) * DI;
    #pragma unroll
    for (int e = 0; e < 16; ++e) {
      int kk = wmma_k16(e, laneHi);
      b0[e] = wr[kk];
      b1[e] = wr[kk + 32];
    }
    v8f acc = {};
    acc = __builtin_amdgcn_wmma_f32_16x16x32_f16(false, a0, false, b0, (short)0, acc, false, false);
    acc = __builtin_amdgcn_wmma_f32_16x16x32_f16(false, a1, false, b1, (short)0, acc, false, false);
    #pragma unroll
    for (int r = 0; r < 8; ++r) {
      int row = row0 + r + laneHi * 8;
      int cc = n0 + ll;
      float sv = s[(size_t)row * CH + cc];
      float val = acc[r] + sv + vskip * sv;   // x + ss2d(ln x) + vss_skip*x
      prev[(size_t)row * CH + cc] = val;
      x_cat[(size_t)row * CIN + chunk * CH + cc] = val;
    }
  }
}

// 9) final: LN128(cvm_skip * x_sh + x_cat)
__global__ void __launch_bounds__(128) k_final(
    const float* __restrict__ x_sh, const float* __restrict__ x_cat,
    const float* __restrict__ cvm_skip, const float* __restrict__ fw,
    const float* __restrict__ fb, float* __restrict__ out) {
  __shared__ float red[128];
  int p = blockIdx.x, c = threadIdx.x;
  float v = cvm_skip[0] * x_sh[(size_t)p * CIN + c] + x_cat[(size_t)p * CIN + c];
  red[c] = v; __syncthreads();
  for (int off = 64; off > 0; off >>= 1) { if (c < off) red[c] += red[c + off]; __syncthreads(); }
  float m = red[0] * (1.f / CIN); __syncthreads();
  float df = v - m;
  red[c] = df * df; __syncthreads();
  for (int off = 64; off > 0; off >>= 1) { if (c < off) red[c] += red[c + off]; __syncthreads(); }
  float rs = rsqrtf(red[0] * (1.f / CIN) + 1e-5f);
  out[(size_t)p * CIN + c] = df * rs * fw[c] + fb[c];
}

// ========================================================================
extern "C" void kernel_launch(void* const* d_in, const int* in_sizes, int n_in,
                              void* d_out, int out_size, void* d_ws, size_t ws_size,
                              hipStream_t stream) {
  const float* x       = (const float*)d_in[0];
  const float* ln1_w   = (const float*)d_in[1];
  const float* ln1_b   = (const float*)d_in[2];
  const float* Wip     = (const float*)d_in[3];
  const float* conv_w  = (const float*)d_in[4];
  const float* conv_b  = (const float*)d_in[5];
  const float* Wxp     = (const float*)d_in[6];
  const float* Wdt     = (const float*)d_in[7];
  const float* bdt     = (const float*)d_in[8];
  const float* A_logs  = (const float*)d_in[9];
  const float* Dsv     = (const float*)d_in[10];
  const float* onw     = (const float*)d_in[11];
  const float* onb     = (const float*)d_in[12];
  const float* Wop     = (const float*)d_in[13];
  const float* vss     = (const float*)d_in[14];
  const float* cvm     = (const float*)d_in[15];
  const float* fw      = (const float*)d_in[16];
  const float* fb      = (const float*)d_in[17];
  float* out = (float*)d_out;

  float* ws = (float*)d_ws;
  size_t off = 0;
  float* x_sh  = ws + off; off += (size_t)BL * CIN;
  float* x_cat = ws + off; off += (size_t)BL * CIN;
  float* s_buf = ws + off; off += (size_t)BL * CH;
  float* prev  = ws + off; off += (size_t)BL * CH;
  float* xz    = ws + off; off += (size_t)BL * 2 * DI;
  float* xxc   = ws + off; off += (size_t)BL * DI;
  float* dtr   = ws + off; off += (size_t)BATCH * KDIR * LSEQ * DTRANK;
  float* Bs    = ws + off; off += (size_t)BATCH * KDIR * LSEQ * DSTATE;
  float* Cs    = ws + off; off += (size_t)BATCH * KDIR * LSEQ * DSTATE;
  float* dts   = ws + off; off += (size_t)BATCH * KDIR * LSEQ * DI;
  float* ys    = ws + off; off += (size_t)BATCH * KDIR * LSEQ * DI;
  float* yact  = ws + off; off += (size_t)BL * DI;
  float* scanP = ws + off; off += (size_t)BATCH * KDIR * NCH * STATES;
  float* scanS = ws + off; off += (size_t)BATCH * KDIR * NCH * STATES;
  float* scanI = ws + off; off += (size_t)BATCH * KDIR * NCH * STATES;
  _Float16* Wip_h = (_Float16*)(ws + off); off += (size_t)(CIN * CH + 1) / 2;
  _Float16* Wxp_h = (_Float16*)(ws + off); off += (size_t)(KDIR * CPAD * DI + 1) / 2;
  _Float16* Wop_h = (_Float16*)(ws + off); off += (size_t)(CH * DI + 1) / 2;

  k_shuffle<<<(BL * CIN + 255) / 256, 256, 0, stream>>>(x, x_sh);
  k_prep_weights<<<(KDIR * CPAD * DI + 255) / 256, 256, 0, stream>>>(
      Wip, Wxp, Wop, Wip_h, Wxp_h, Wop_h);

  for (int i = 0; i < 4; ++i) {
    k_chunk_input<<<(BL * CH + 255) / 256, 256, 0, stream>>>(x_sh, prev, s_buf, i, i == 0);
    k_ln_inproj<<<BL / 16, 256, 0, stream>>>(s_buf, ln1_w, ln1_b, Wip_h, xz);
    k_conv_silu<<<(BL * DI + 255) / 256, 256, 0, stream>>>(xz, conv_w, conv_b, xxc);
    k_xproj<<<BATCH * KDIR * (LSEQ / 16), 32, 0, stream>>>(xxc, Wxp_h, dtr, Bs, Cs);
    k_dtproj<<<(BATCH * KDIR * LSEQ * DI + 255) / 256, 256, 0, stream>>>(dtr, Wdt, bdt, dts);
    k_scan_chunk<<<BATCH * KDIR * NCH, 1024, 0, stream>>>(xxc, dts, Bs, A_logs, scanP, scanS);
    k_scan_prefix<<<BATCH * KDIR, 1024, 0, stream>>>(scanP, scanS, scanI);
    k_scan_apply<<<BATCH * KDIR * NCH, 1024, 0, stream>>>(xxc, dts, Bs, Cs, A_logs, Dsv,
                                                          scanI, ys);
    k_combine<<<BL, 64, 0, stream>>>(ys, xz, onw, onb, yact);
    k_outproj<<<BL / 16, 32, 0, stream>>>(yact, Wop_h, s_buf, vss, prev, x_cat, i);
  }

  k_final<<<BL, 128, 0, stream>>>(x_sh, x_cat, cvm, fw, fb, out);
}
